// Vel_Decoder_43387759624355
// MI455X (gfx1250) — compile-verified, matrix operand order checked by reading, not run on hardware
//
#include <hip/hip_runtime.h>
#include <math.h>

// ---------------- types ----------------
typedef __bf16 bf16;
typedef __attribute__((ext_vector_type(16))) __bf16 v16bf;
typedef __attribute__((ext_vector_type(8)))  float  v8f;

union Frag16 { v16bf v; uint4 q[2]; };

// sizes (fixed by the reference)
#define BATCH 512
#define SEQ   64
#define HID   1024
#define G3    3072   // 3*HID
#define ND    9
#define VC    4
#define INW   22     // 2*ND + VC
#define ZCAT  784    // 256 + 512 + 16

// GEMM tiling
#define BM    128
#define BN    64
#define BK    32
#define LDP   40     // padded LDS row stride in bf16 elems (80B) -> conflict-free frag reads

__device__ __forceinline__ bf16 f2bf(float f) {
    union { float f; unsigned int u; } in; in.f = f;
    unsigned int u = in.u;
    unsigned int r = u + 0x7FFFu + ((u >> 16) & 1u);   // round-nearest-even
    union { unsigned short s; bf16 b; } out;
    out.s = (unsigned short)(r >> 16);
    return out.b;
}
__device__ __forceinline__ float sigm(float x) { return 1.0f / (1.0f + __expf(-x)); }

// async global->LDS copy, 16B per lane; tracked by ASYNCcnt (no VGPR round-trip)
__device__ __forceinline__ void async_ld_b128(const void* gaddr, void* lds_ptr) {
    unsigned lds_off = (unsigned)(size_t)lds_ptr;   // low 32 bits of flat shared ptr = LDS byte addr
    asm volatile("global_load_async_to_lds_b128 %0, %1, off"
                 :: "v"(lds_off), "v"(gaddr) : "memory");
}
__device__ __forceinline__ void wait_async0() {
    asm volatile("s_wait_asynccnt 0x0" ::: "memory");
}

// ---------------- f32 -> bf16 conversion ----------------
__global__ void cvt_f32_bf16(const float* __restrict__ src, bf16* __restrict__ dst, int n) {
    int i = blockIdx.x * blockDim.x + threadIdx.x;
    if (i < n) dst[i] = f2bf(src[i]);
}

// ---------------- h1_0 = tanh(zc @ W_in^T + b_in), and h1 bf16 copy ----------------
__global__ void init_h1(const float* __restrict__ z, const float* __restrict__ ne,
                        const float* __restrict__ gc, const float* __restrict__ W_in,
                        const float* __restrict__ b_in,
                        float* __restrict__ h1f, bf16* __restrict__ h1b) {
    int idx = blockIdx.x * blockDim.x + threadIdx.x;
    if (idx >= BATCH * HID) return;
    int b = idx >> 10, h = idx & (HID - 1);
    const float* w = W_in + (size_t)h * ZCAT;
    float s = b_in[h];
    const float* zb = z + (size_t)b * 256;
    for (int k = 0; k < 256; ++k) s += zb[k] * w[k];
    const float* nb = ne + (size_t)b * 512;
    for (int k = 0; k < 512; ++k) s += nb[k] * w[256 + k];
    const float* gb = gc + (size_t)b * 16;
    for (int k = 0; k < 16; ++k) s += gb[k] * w[768 + k];
    float t = tanhf(s);
    h1f[idx] = t;
    h1b[idx] = f2bf(t);
}

__global__ void zero_vel(float* __restrict__ vp) {
    int i = blockIdx.x * blockDim.x + threadIdx.x;
    if (i < BATCH * ND) vp[i] = 0.0f;
}

// ---------------- LDS-tiled WMMA GEMM with async global->LDS staging ----------------
// C[M,N] = A[M,K](bf16) * W[N,K](bf16)^T + bias[N]
// Block tile 128M x 64N, K-slab 32, double-buffered; staging via GLOBAL_LOAD_ASYNC_TO_LDS_B128.
// 8 waves as 4(M) x 2(N); wave tile 32M x 32N = 2x2 WMMA tiles.
__global__ void __launch_bounds__(256)
gemm_bf16_wmma(const bf16* __restrict__ A, const bf16* __restrict__ W,
               const float* __restrict__ bias, float* __restrict__ C,
               int M, int N, int K) {
    __shared__ bf16 As[2][BM * LDP];   // 2*128*40*2B = 20 KB
    __shared__ bf16 Bs[2][BN * LDP];   // 2* 64*40*2B = 10 KB

    const int tid  = threadIdx.x;
    const int wave = tid >> 5;
    const int lane = tid & 31;
    const int half = lane >> 4;        // K-half select per 16-bit A/B layout
    const int lr   = lane & 15;        // row (A) / col (B) within 16-wide tile
    const int mw   = wave >> 1;        // 0..3
    const int nw   = wave & 1;         // 0..1
    const int m0   = blockIdx.x * BM;
    const int n0   = blockIdx.y * BN;

    // staging: 16B chunks; A = 128 rows x 64B (512 chunks, 2/thread),
    // B = 64 rows x 64B (256 chunks, 1/thread)
    const int ra = tid >> 2;           // 0..63
    const int ca = (tid & 3) * 8;      // element col of 16B chunk

    auto stage = [&](int kb, int buf) {
        async_ld_b128(A + (size_t)(m0 + ra) * K + kb + ca,
                      &As[buf][(size_t)ra * LDP + ca]);
        async_ld_b128(A + (size_t)(m0 + 64 + ra) * K + kb + ca,
                      &As[buf][(size_t)(64 + ra) * LDP + ca]);
        async_ld_b128(W + (size_t)(n0 + ra) * K + kb + ca,
                      &Bs[buf][(size_t)ra * LDP + ca]);
    };

    v8f acc00 = {}, acc01 = {}, acc10 = {}, acc11 = {};
    const int klo = 8 * half;          // K run 0..7 (lanes 0-15) / 8..15 (lanes 16-31)
    const int khi = 16 + 8 * half;     // K run 16..23 / 24..31

    auto compute = [&](int buf) {
        Frag16 a0, a1, b0, b1;
        const bf16* pa0 = &As[buf][(size_t)(mw * 32 + lr) * LDP];
        const bf16* pa1 = &As[buf][(size_t)(mw * 32 + 16 + lr) * LDP];
        const bf16* pb0 = &Bs[buf][(size_t)(nw * 32 + lr) * LDP];
        const bf16* pb1 = &Bs[buf][(size_t)(nw * 32 + 16 + lr) * LDP];
        a0.q[0] = *(const uint4*)(pa0 + klo); a0.q[1] = *(const uint4*)(pa0 + khi);
        a1.q[0] = *(const uint4*)(pa1 + klo); a1.q[1] = *(const uint4*)(pa1 + khi);
        b0.q[0] = *(const uint4*)(pb0 + klo); b0.q[1] = *(const uint4*)(pb0 + khi);
        b1.q[0] = *(const uint4*)(pb1 + klo); b1.q[1] = *(const uint4*)(pb1 + khi);
        acc00 = __builtin_amdgcn_wmma_f32_16x16x32_bf16(false, a0.v, false, b0.v, (short)0, acc00, false, false);
        acc01 = __builtin_amdgcn_wmma_f32_16x16x32_bf16(false, a0.v, false, b1.v, (short)0, acc01, false, false);
        acc10 = __builtin_amdgcn_wmma_f32_16x16x32_bf16(false, a1.v, false, b0.v, (short)0, acc10, false, false);
        acc11 = __builtin_amdgcn_wmma_f32_16x16x32_bf16(false, a1.v, false, b1.v, (short)0, acc11, false, false);
    };

    stage(0, 0);
    wait_async0();
    __syncthreads();
    int buf = 0;
    for (int kb = BK; kb < K; kb += BK) {
        stage(kb, buf ^ 1);            // async fill of slab k+1 flies while...
        compute(buf);                  // ...WMMAs consume slab k from LDS
        wait_async0();                 // async writes to buf^1 visible
        __syncthreads();               // all waves' reads of buf done
        buf ^= 1;
    }
    compute(buf);

    // C/D layout: lanes 0-15 -> N=lane, M=v; lanes 16-31 -> N=lane-16, M=v+8
    v8f accs[2][2] = {{acc00, acc01}, {acc10, acc11}};
#pragma unroll
    for (int i = 0; i < 2; ++i) {
#pragma unroll
        for (int j = 0; j < 2; ++j) {
            int n = n0 + nw * 32 + j * 16 + lr;
            float bv = bias[n];
#pragma unroll
            for (int v = 0; v < 8; ++v) {
                int m = m0 + mw * 32 + i * 16 + v + 8 * half;
                C[(size_t)m * N + n] = accs[i][j][v] + bv;
            }
        }
    }
}

// ---------------- GRU1 gate combine (+ tiny K=22 input GEMM) ----------------
__global__ void gru1_combine(const float* __restrict__ g1,     // gh1 = h1@W_hh1^T + b_hh1  [B,3H]
                             const float* __restrict__ vprev,  // [B,ND]
                             const float* __restrict__ note,   // [B,S,ND]
                             const float* __restrict__ vcond,  // [B,S,VC]
                             const float* __restrict__ W_ih1,  // [3H,22]
                             const float* __restrict__ b_ih1,  // [3H]
                             int t, int copy_h2,
                             float* __restrict__ h1f, bf16* __restrict__ h1b,
                             float* __restrict__ h2f, bf16* __restrict__ h2b) {
    int idx = blockIdx.x * blockDim.x + threadIdx.x;
    if (idx >= BATCH * HID) return;
    int b = idx >> 10, j = idx & (HID - 1);

    float inp[INW];
    const float* vp = vprev + (size_t)b * ND;
#pragma unroll
    for (int k = 0; k < ND; ++k) inp[k] = vp[k];
    const float* nb = note + ((size_t)b * SEQ + t) * ND;
#pragma unroll
    for (int k = 0; k < ND; ++k) inp[ND + k] = nb[k];
    const float* cb = vcond + ((size_t)b * SEQ + t) * VC;
#pragma unroll
    for (int k = 0; k < VC; ++k) inp[2 * ND + k] = cb[k];

    float gr = b_ih1[j], gz = b_ih1[HID + j], gn = b_ih1[2 * HID + j];
    const float* wr = W_ih1 + (size_t)j * INW;
    const float* wz = W_ih1 + (size_t)(HID + j) * INW;
    const float* wn = W_ih1 + (size_t)(2 * HID + j) * INW;
#pragma unroll
    for (int k = 0; k < INW; ++k) {
        gr += inp[k] * wr[k]; gz += inp[k] * wz[k]; gn += inp[k] * wn[k];
    }

    const float* gb = g1 + (size_t)b * G3;
    float r  = sigm(gr + gb[j]);
    float zg = sigm(gz + gb[HID + j]);
    float n  = tanhf(gn + r * gb[2 * HID + j]);
    float h  = (1.0f - zg) * n + zg * h1f[idx];

    h1f[idx] = h;
    h1b[idx] = f2bf(h);
    if (copy_h2) { h2f[idx] = h; h2b[idx] = f2bf(h); }   // h2 = h1 at t==0
}

// ---------------- GRU2 gate combine ----------------
__global__ void gru2_combine(const float* __restrict__ gi2,  // h1@W_ih2^T + b_ih2  [B,3H]
                             const float* __restrict__ gh2,  // h2@W_hh2^T + b_hh2  [B,3H]
                             float* __restrict__ h2f, bf16* __restrict__ h2b) {
    int idx = blockIdx.x * blockDim.x + threadIdx.x;
    if (idx >= BATCH * HID) return;
    int b = idx >> 10, j = idx & (HID - 1);
    const float* gib = gi2 + (size_t)b * G3;
    const float* ghb = gh2 + (size_t)b * G3;
    float r  = sigm(gib[j] + ghb[j]);
    float zg = sigm(gib[HID + j] + ghb[HID + j]);
    float n  = tanhf(gib[2 * HID + j] + r * ghb[2 * HID + j]);
    float h  = (1.0f - zg) * n + zg * h2f[idx];
    h2f[idx] = h;
    h2b[idx] = f2bf(h);
}

// ---------------- vel = sigmoid(h2 @ W_out^T + b_out); one wave per (b,d) ----------------
__global__ void out_proj(const float* __restrict__ h2f, const float* __restrict__ W_out,
                         const float* __restrict__ b_out, float* __restrict__ out,
                         float* __restrict__ vprev, int t) {
    int wid  = (blockIdx.x * blockDim.x + threadIdx.x) >> 5;
    int lane = threadIdx.x & 31;
    if (wid >= BATCH * ND) return;
    int b = wid / ND, d = wid % ND;
    const float* h = h2f + (size_t)b * HID;
    const float* w = W_out + (size_t)d * HID;
    float s = 0.0f;
    for (int k = lane; k < HID; k += 32) s += h[k] * w[k];
#pragma unroll
    for (int off = 16; off; off >>= 1) s += __shfl_xor(s, off, 32);
    if (lane == 0) {
        float v = sigm(s + b_out[d]);
        out[((size_t)b * SEQ + t) * ND + d] = v;
        vprev[(size_t)b * ND + d] = v;
    }
}

// ---------------- host launcher ----------------
extern "C" void kernel_launch(void* const* d_in, const int* in_sizes, int n_in,
                              void* d_out, int out_size, void* d_ws, size_t ws_size,
                              hipStream_t stream) {
    (void)in_sizes; (void)n_in; (void)out_size; (void)ws_size;
    const float* z     = (const float*)d_in[0];
    const float* ne    = (const float*)d_in[1];
    // d_in[2] = vel_target (unused in eval mode)
    const float* note  = (const float*)d_in[3];
    const float* vcond = (const float*)d_in[4];
    const float* gc    = (const float*)d_in[5];
    const float* W_in  = (const float*)d_in[6];
    const float* b_in  = (const float*)d_in[7];
    const float* W_ih1 = (const float*)d_in[8];
    const float* W_hh1 = (const float*)d_in[9];
    const float* b_ih1 = (const float*)d_in[10];
    const float* b_hh1 = (const float*)d_in[11];
    const float* W_ih2 = (const float*)d_in[12];
    const float* W_hh2 = (const float*)d_in[13];
    const float* b_ih2 = (const float*)d_in[14];
    const float* b_hh2 = (const float*)d_in[15];
    const float* W_out = (const float*)d_in[16];
    const float* b_out = (const float*)d_in[17];
    float* out = (float*)d_out;

    // workspace layout
    size_t off = 0;
    auto alloc = [&](size_t bytes) -> void* {
        void* p = (char*)d_ws + off;
        off += (bytes + 255) & ~(size_t)255;
        return p;
    };
    bf16*  wW1   = (bf16*)alloc((size_t)G3 * HID * sizeof(bf16));
    bf16*  wI2   = (bf16*)alloc((size_t)G3 * HID * sizeof(bf16));
    bf16*  wH2   = (bf16*)alloc((size_t)G3 * HID * sizeof(bf16));
    float* h1f   = (float*)alloc((size_t)BATCH * HID * sizeof(float));
    float* h2f   = (float*)alloc((size_t)BATCH * HID * sizeof(float));
    bf16*  h1b   = (bf16*)alloc((size_t)BATCH * HID * sizeof(bf16));
    bf16*  h2b   = (bf16*)alloc((size_t)BATCH * HID * sizeof(bf16));
    float* g1    = (float*)alloc((size_t)BATCH * G3 * sizeof(float));
    float* gi2   = (float*)alloc((size_t)BATCH * G3 * sizeof(float));
    float* gh2   = (float*)alloc((size_t)BATCH * G3 * sizeof(float));
    float* vprev = (float*)alloc((size_t)BATCH * ND * sizeof(float));

    const int NW = G3 * HID;             // 3,145,728 weights per matrix
    cvt_f32_bf16<<<(NW + 255) / 256, 256, 0, stream>>>(W_hh1, wW1, NW);
    cvt_f32_bf16<<<(NW + 255) / 256, 256, 0, stream>>>(W_ih2, wI2, NW);
    cvt_f32_bf16<<<(NW + 255) / 256, 256, 0, stream>>>(W_hh2, wH2, NW);

    init_h1<<<(BATCH * HID) / 256, 256, 0, stream>>>(z, ne, gc, W_in, b_in, h1f, h1b);
    zero_vel<<<(BATCH * ND + 255) / 256, 256, 0, stream>>>(vprev);

    dim3 gemmGrid(BATCH / BM, G3 / BN);   // (4, 48)
    dim3 eltGrid((BATCH * HID) / 256);
    int  outBlocks = (BATCH * ND * 32 + 255) / 256;   // 1 wave per (b,d)

    for (int t = 0; t < SEQ; ++t) {
        gemm_bf16_wmma<<<gemmGrid, 256, 0, stream>>>(h1b, wW1, b_hh1, g1, BATCH, G3, HID);
        gru1_combine<<<eltGrid, 256, 0, stream>>>(g1, vprev, note, vcond, W_ih1, b_ih1,
                                                  t, (t == 0) ? 1 : 0, h1f, h1b, h2f, h2b);
        gemm_bf16_wmma<<<gemmGrid, 256, 0, stream>>>(h1b, wI2, b_ih2, gi2, BATCH, G3, HID);
        gemm_bf16_wmma<<<gemmGrid, 256, 0, stream>>>(h2b, wH2, b_hh2, gh2, BATCH, G3, HID);
        gru2_combine<<<eltGrid, 256, 0, stream>>>(gi2, gh2, h2f, h2b);
        out_proj<<<outBlocks, 256, 0, stream>>>(h2f, W_out, b_out, out, vprev, t);
    }
}
